// ThreeGraphGAT_82575041232969
// MI455X (gfx1250) — matrix operand order repaired and checked by reading, not compiled, not run on hardware
//
#include <hip/hip_runtime.h>
#include <hip/hip_bf16.h>
#include <cstdint>

#define NNODES 100000
#define NEDGES 3200000
#define DF 32
#define NEG_SLOPE 0.2f

typedef __attribute__((ext_vector_type(16))) _Float16 v16h;
typedef __attribute__((ext_vector_type(8)))  float    v8f;

// ---------- small dense projections (K = 3 / 11 / 14) ----------
__global__ void __launch_bounds__(256)
proj_kernel(const float* __restrict__ in1, int d1,
            const float* __restrict__ in2, int d2,
            const float* __restrict__ W, const float* __restrict__ b,
            _Float16* __restrict__ out, int n)
{
    int tid = blockIdx.x * blockDim.x + threadIdx.x;
    if (tid >= n * DF) return;
    int node = tid >> 5, j = tid & 31;
    float acc = b[j];
    for (int k = 0; k < d1; ++k) acc += in1[node * d1 + k] * W[k * DF + j];
    if (in2)
        for (int k = 0; k < d2; ++k) acc += in2[node * d2 + k] * W[(d1 + k) * DF + j];
    out[node * DF + j] = (_Float16)acc;
}

// ---------- transpose + f32->f16 weight convert: W[K,N] -> Wt[N,K] ----------
__global__ void __launch_bounds__(256)
wcvt_kernel(const float* __restrict__ W, _Float16* __restrict__ Wt, int K, int N)
{
    int t = blockIdx.x * blockDim.x + threadIdx.x;
    if (t >= K * N) return;
    int k = t / N, n = t % N;
    Wt[n * K + k] = (_Float16)W[k * N + n];
}

// ---------- h = f @ Wg via v_wmma_f32_16x16x32_f16 (WgT: f16 [32,32]) ----------
__global__ void __launch_bounds__(256)
gemm_h_kernel(const _Float16* __restrict__ f, const _Float16* __restrict__ WgT,
              float* __restrict__ h, int ntiles)
{
    int lane = threadIdx.x & 31;
    int wave = threadIdx.x >> 5;
    int tile = blockIdx.x * 8 + wave;
    if (tile >= ntiles) tile = ntiles - 1;   // clamp: EXEC stays all-ones for WMMA
    int row0 = tile * 16;
    int mrow = lane & 15;
    int hi = lane >> 4;

    // A 16x32 f16: lanes 0-15 hold K 0-15, lanes 16-31 hold K 16-31
    v16h a;
    const _Float16* frow = f + (size_t)(row0 + mrow) * DF;
#pragma unroll
    for (int j = 0; j < 8; ++j) a[j] = frow[hi * 16 + j];
#pragma unroll
    for (int j = 0; j < 8; ++j) a[8 + j] = frow[hi * 16 + 8 + j];

#pragma unroll
    for (int nt = 0; nt < 2; ++nt) {
        // B 32x16: lane holds col = N, contiguous K run from transposed f16 weights
        int col = nt * 16 + mrow;
        v16h bm = *(const v16h*)(WgT + col * 32 + hi * 16);
        v8f c = {};
        c = __builtin_amdgcn_wmma_f32_16x16x32_f16(false, a, false, bm, (short)0, c, false, false);
#pragma unroll
        for (int v = 0; v < 8; ++v) {
            int m = v + hi * 8;
            h[(size_t)(row0 + m) * DF + col] = c[v];
        }
    }
}

// ---------- per-node attention logits a_s, a_d ----------
__global__ void __launch_bounds__(256)
att_kernel(const float* __restrict__ h, const float* __restrict__ asrc,
           const float* __restrict__ adst, float* __restrict__ a_s,
           float* __restrict__ a_d, int n)
{
    int tid = blockIdx.x * blockDim.x + threadIdx.x;
    if (tid >= n * 4) return;
    int node = tid >> 2, hd = tid & 3;
    const float* hp = h + (size_t)node * DF + hd * 8;
    float s = 0.f, d = 0.f;
#pragma unroll
    for (int c = 0; c < 8; ++c) { s += hp[c] * asrc[hd * 8 + c]; d += hp[c] * adst[hd * 8 + c]; }
    a_s[tid] = s;
    a_d[tid] = d;
}

__global__ void __launch_bounds__(256)
fill_u32(uint32_t* __restrict__ p, uint32_t v, int n)
{
    int t = blockIdx.x * blockDim.x + threadIdx.x;
    if (t < n) p[t] = v;
}

// ---------- single edge pass: shift-invariant softmax accumulation ----------
// den[dst,h] += exp(e); num[dst,h,:] += h[src,h,:]*exp(e)   (e clamped to +-60)
__global__ void __launch_bounds__(256)
edge_accum_kernel(const int* __restrict__ ei, const float* __restrict__ a_s,
                  const float* __restrict__ a_d, const float* __restrict__ hbuf,
                  float* __restrict__ den, float* __restrict__ num, int etot)
{
    int t = blockIdx.x * blockDim.x + threadIdx.x;
    if (t >= etot * 4) return;
    int eidx = t >> 2, hd = t & 3;
    int s, d;
    if (eidx < NEDGES) { s = ei[eidx]; d = ei[NEDGES + eidx]; }  // row0=src, row1=dst
    else               { s = d = eidx - NEDGES; }                // implicit self loops
    float e = a_s[s * 4 + hd] + a_d[d * 4 + hd];
    e = e > 0.f ? e : NEG_SLOPE * e;                             // leaky_relu
    e = fminf(fmaxf(e, -60.f), 60.f);                            // fp32-safe range
    float ex = __expf(e);
    atomicAdd(&den[d * 4 + hd], ex);                             // L2-resident atomics
    const float* hs = hbuf + (size_t)s * DF + hd * 8;            // L2-resident gather
    float* nd = num + (size_t)d * DF + hd * 8;
#pragma unroll
    for (int c = 0; c < 8; ++c) atomicAdd(&nd[c], hs[c] * ex);
}

// ---------- normalize + bias + ELU, pack into Hcat columns ----------
__global__ void __launch_bounds__(256)
finalize_kernel(const float* __restrict__ num, const float* __restrict__ den,
                const float* __restrict__ bg, _Float16* __restrict__ Hcat,
                int goff, int n)
{
    int t = blockIdx.x * blockDim.x + threadIdx.x;
    if (t >= n * DF) return;
    int node = t >> 5, j = t & 31, hd = j >> 3;
    float v = num[t] / den[node * 4 + hd] + bg[j];
    v = v > 0.f ? v : (__expf(v) - 1.f);                         // ELU
    Hcat[(size_t)node * 96 + goff + j] = (_Float16)v;
}

// ---------- fuse MLP: relu(Hcat@W1+b1)@W2+b2, chained WMMA with LDS relayout ----
__global__ void __launch_bounds__(256)
fuse_kernel(const _Float16* __restrict__ Hcat, const _Float16* __restrict__ W1T,
            const float* __restrict__ b1, const _Float16* __restrict__ W2T,
            const float* __restrict__ b2, float* __restrict__ out, int ntiles)
{
    __shared__ _Float16 lds[8][16 * DF];    // per-wave 16x32 staging tile
    int lane = threadIdx.x & 31;
    int wave = threadIdx.x >> 5;
    int tile = blockIdx.x * 8 + wave;
    int vt = tile < ntiles ? tile : ntiles - 1;   // clamp keeps EXEC all-ones
    int row0 = vt * 16;
    int mrow = lane & 15, hi = lane >> 4;

    v8f acc0 = {}, acc1 = {};
    const _Float16* hrow = Hcat + (size_t)(row0 + mrow) * 96;
#pragma unroll
    for (int q = 0; q < 3; ++q) {           // K = 96 = 3 x 32
        v16h a;
#pragma unroll
        for (int j = 0; j < 8; ++j) a[j] = hrow[q * 32 + hi * 16 + j];
#pragma unroll
        for (int j = 0; j < 8; ++j) a[8 + j] = hrow[q * 32 + hi * 16 + 8 + j];
        v16h bm0 = *(const v16h*)(W1T + (mrow) * 96 + q * 32 + hi * 16);
        v16h bm1 = *(const v16h*)(W1T + (16 + mrow) * 96 + q * 32 + hi * 16);
        acc0 = __builtin_amdgcn_wmma_f32_16x16x32_f16(false, a, false, bm0, (short)0, acc0, false, false);
        acc1 = __builtin_amdgcn_wmma_f32_16x16x32_f16(false, a, false, bm1, (short)0, acc1, false, false);
    }
    // bias + relu, D-layout -> row-major f16 LDS
#pragma unroll
    for (int v = 0; v < 8; ++v) {
        int mm = v + hi * 8;
        float t0 = acc0[v] + b1[mrow];          t0 = t0 > 0.f ? t0 : 0.f;
        float t1 = acc1[v] + b1[16 + mrow];     t1 = t1 > 0.f ? t1 : 0.f;
        lds[wave][mm * DF + mrow] = (_Float16)t0;
        lds[wave][mm * DF + 16 + mrow] = (_Float16)t1;
    }
    asm volatile("s_wait_dscnt 0" ::: "memory");

    // reload in A layout (per-wave private region; in-order DS unit)
    v16h a2;
    const _Float16* lr = &lds[wave][mrow * DF];
#pragma unroll
    for (int j = 0; j < 8; ++j) a2[j] = lr[hi * 16 + j];
#pragma unroll
    for (int j = 0; j < 8; ++j) a2[8 + j] = lr[hi * 16 + 8 + j];

#pragma unroll
    for (int nt = 0; nt < 2; ++nt) {
        int col = nt * 16 + mrow;
        v16h bm = *(const v16h*)(W2T + col * 32 + hi * 16);
        v8f c = {};
        c = __builtin_amdgcn_wmma_f32_16x16x32_f16(false, a2, false, bm, (short)0, c, false, false);
        if (tile < ntiles) {
#pragma unroll
            for (int v = 0; v < 8; ++v) {
                int mm = v + hi * 8;
                out[(size_t)(row0 + mm) * DF + col] = c[v] + b2[col];
            }
        }
    }
}

extern "C" void kernel_launch(void* const* d_in, const int* in_sizes, int n_in,
                              void* d_out, int out_size, void* d_ws, size_t ws_size,
                              hipStream_t stream)
{
    const float* F_p = (const float*)d_in[0];
    const float* F_a = (const float*)d_in[1];
    const int* edges[3] = { (const int*)d_in[2], (const int*)d_in[3], (const int*)d_in[4] };
    const float* Wp = (const float*)d_in[5];  const float* bp = (const float*)d_in[6];
    const float* Wa = (const float*)d_in[7];  const float* ba = (const float*)d_in[8];
    const float* Ws = (const float*)d_in[9];  const float* bs = (const float*)d_in[10];
    const float* W1 = (const float*)d_in[23]; const float* b1 = (const float*)d_in[24];
    const float* W2 = (const float*)d_in[25]; const float* b2 = (const float*)d_in[26];
    float* out = (float*)d_out;

    // ---- carve workspace ----
    char* ws = (char*)d_ws;
    size_t off = 0;
    auto carve = [&](size_t bytes) { void* p = ws + off; off += (bytes + 255) & ~(size_t)255; return p; };
    _Float16* Hcat = (_Float16*)carve((size_t)NNODES * 96 * 2);
    _Float16* fbuf = (_Float16*)carve((size_t)NNODES * 32 * 2);
    float* hbuf = (float*)carve((size_t)NNODES * 32 * 4);
    float* a_s  = (float*)carve((size_t)NNODES * 4 * 4);
    float* a_d  = (float*)carve((size_t)NNODES * 4 * 4);
    float* den  = (float*)carve((size_t)NNODES * 4 * 4);
    float* num  = (float*)carve((size_t)NNODES * 32 * 4);
    _Float16* WgT = (_Float16*)carve(32 * 32 * 2);
    _Float16* W1T = (_Float16*)carve(96 * 32 * 2);
    _Float16* W2T = (_Float16*)carve(32 * 32 * 2);
    (void)ws_size; (void)in_sizes; (void)n_in; (void)out_size;

    const int etot = NEDGES + NNODES;         // edges + self loops
    const int ntiles = NNODES / 16;           // 6250, exact
    const int gNodeF = (NNODES * 32 + 255) / 256;
    const int gNodeH = (NNODES * 4 + 255) / 256;
    const int gTile  = (ntiles + 7) / 8;
    const int gEdgeH = (etot * 4 + 255) / 256;

    // one-time weight transposes for coalesced WMMA B fragments
    wcvt_kernel<<<(96 * 32 + 255) / 256, 256, 0, stream>>>(W1, W1T, 96, 32);
    wcvt_kernel<<<(32 * 32 + 255) / 256, 256, 0, stream>>>(W2, W2T, 32, 32);

    for (int g = 0; g < 3; ++g) {
        if (g == 0)
            proj_kernel<<<gNodeF, 256, 0, stream>>>(F_p, 3, F_a, 11, Ws, bs, fbuf, NNODES);
        else if (g == 1)
            proj_kernel<<<gNodeF, 256, 0, stream>>>(F_p, 3, nullptr, 0, Wp, bp, fbuf, NNODES);
        else
            proj_kernel<<<gNodeF, 256, 0, stream>>>(F_a, 11, nullptr, 0, Wa, ba, fbuf, NNODES);

        const float* Wg   = (const float*)d_in[11 + 4 * g];
        const float* asrc = (const float*)d_in[12 + 4 * g];
        const float* adst = (const float*)d_in[13 + 4 * g];
        const float* bg   = (const float*)d_in[14 + 4 * g];

        wcvt_kernel<<<(32 * 32 + 255) / 256, 256, 0, stream>>>(Wg, WgT, 32, 32);
        gemm_h_kernel<<<gTile, 256, 0, stream>>>(fbuf, WgT, hbuf, ntiles);
        att_kernel<<<gNodeH, 256, 0, stream>>>(hbuf, asrc, adst, a_s, a_d, NNODES);

        fill_u32<<<(NNODES * 4 + 255) / 256, 256, 0, stream>>>((uint32_t*)den, 0u, NNODES * 4);
        fill_u32<<<(NNODES * 32 + 255) / 256, 256, 0, stream>>>((uint32_t*)num, 0u, NNODES * 32);

        edge_accum_kernel<<<gEdgeH, 256, 0, stream>>>(edges[g], a_s, a_d, hbuf, den, num, etot);
        finalize_kernel<<<gNodeF, 256, 0, stream>>>(num, den, bg, Hcat, g * 32, NNODES);
    }

    fuse_kernel<<<gTile, 256, 0, stream>>>(Hcat, W1T, b1, W2T, b2, out, ntiles);
}
